// Transformer_24550033064219
// MI455X (gfx1250) — compile-verified, matrix operand order checked by reading, not run on hardware
//
#include <hip/hip_runtime.h>
#include <hip/hip_bf16.h>
#include <math.h>

// ---------------------------------------------------------------------------
// Types
// ---------------------------------------------------------------------------
typedef __bf16 bf16_t;
typedef __attribute__((ext_vector_type(16))) __bf16 v16bf;
typedef __attribute__((ext_vector_type(8)))  float  v8f;

// Model constants
#define BB   8
#define SS   1024
#define DD   768
#define FF   3072
#define CLS  17
#define OUTD 1024
#define NL   12
#define BS   (BB * SS)      // 8192 rows
#define DQKV (3 * DD)       // 2304

// GEMM tiling
#define TILE_M 128
#define TILE_N 128
#define TILE_K 32

// ---------------------------------------------------------------------------
// CDNA5 async global->LDS staging (GLOBAL_LOAD_ASYNC_TO_LDS_B128, ASYNCcnt),
// via inline asm (ISA 15.18.3 op 98; VDST = wave-relative LDS byte address,
// VADDR = 64-bit global VA, saddr = off). The clang builtin exists but its
// pointer parameters use CUDA language address spaces that HIP source cannot
// name portably, so asm is the robust route per the bridge doc.
// ---------------------------------------------------------------------------
#if defined(__HIP_DEVICE_COMPILE__)
#define USE_ASYNC_COPY 1
#else
#define USE_ASYNC_COPY 0
#endif

__device__ __forceinline__ void async_copy_b128(const void* gsrc, void* lds_dst)
{
#if USE_ASYNC_COPY
    // Generic LDS pointer -> wave-relative DS byte address (low 32 bits).
    unsigned           ldsaddr = (unsigned)(size_t)lds_dst;
    unsigned long long gaddr   = (unsigned long long)(size_t)gsrc;
    asm volatile("global_load_async_to_lds_b128 %0, %1, off"
                 :
                 : "v"(ldsaddr), "v"(gaddr)
                 : "memory");
#else
    *reinterpret_cast<float4*>(lds_dst) = *reinterpret_cast<const float4*>(gsrc);
#endif
}

__device__ __forceinline__ void async_stage_wait()
{
#if USE_ASYNC_COPY
    asm volatile("s_wait_asynccnt 0" ::: "memory");
#endif
}

// ---------------------------------------------------------------------------
// Tiled WMMA GEMM:  C[M,N] = A[M,K] (bf16) * Bt[N,K]^T (bf16)  (+bias)(+resid)
// 256 threads = 8 wave32 waves in a 2(M) x 4(N) grid; each wave does 64x32
// of C via 4x2 v_wmma_f32_16x16x32_bf16 fragments. Double-buffered LDS.
// ---------------------------------------------------------------------------
union Frag {
    v16bf  v;
    float4 f4[2];
};

__device__ __forceinline__ void load_tile_g2l(const bf16_t* __restrict__ g, int ld,
                                              int row0, int k0,
                                              bf16_t* __restrict__ sm, int tid)
{
    // 128 rows x 32 cols bf16 = 8KB; 256 threads x 2 iters x 16B
#pragma unroll
    for (int it = 0; it < 2; ++it) {
        int idx = tid + it * 256;        // 0..511
        int r   = idx >> 2;              // 0..127
        int c   = (idx & 3) * 8;         // element offset within K-slice
        const bf16_t* src = g  + (size_t)(row0 + r) * ld + k0 + c;
        bf16_t*       dst = sm + r * TILE_K + c;
        async_copy_b128(src, dst);
    }
}

__device__ __forceinline__ void load_frag_lds(const bf16_t* __restrict__ sm,
                                              int rbase, int lane, Frag& fr)
{
    // ISA 16-bit A/B layout: lane = row%16; lanes 0-15 hold K 0..7 & 16..23,
    // lanes 16-31 hold K 8..15 & 24..31 (each half = one 16B LDS read).
    int r = rbase + (lane & 15);
    int h = (lane >> 4) * 8;
    fr.f4[0] = *reinterpret_cast<const float4*>(sm + r * TILE_K + h);
    fr.f4[1] = *reinterpret_cast<const float4*>(sm + r * TILE_K + 16 + h);
}

template <int ACT, bool RESID, bool OUTBF>
__global__ __launch_bounds__(256)
void gemm_bt_kernel(const bf16_t* __restrict__ A, const bf16_t* __restrict__ Bt,
                    const float* __restrict__ bias, const float* __restrict__ resid,
                    void* __restrict__ Out,
                    int M, int N, int K,
                    long long sA, long long sB, long long sC)
{
    __shared__ bf16_t As[2][TILE_M * TILE_K];
    __shared__ bf16_t Bs[2][TILE_N * TILE_K];

    const int tid  = threadIdx.x;
    const int lane = tid & 31;
    const int wave = tid >> 5;
    const int wm   = wave >> 2;   // 0..1
    const int wn   = wave & 3;    // 0..3
    const int bm   = blockIdx.y * TILE_M;
    const int bn   = blockIdx.x * TILE_N;
    const int bz   = blockIdx.z;

    const bf16_t* Ab = A  + (size_t)bz * sA;
    const bf16_t* Bb = Bt + (size_t)bz * sB;

    v8f acc[4][2] = {};

    load_tile_g2l(Ab, K, bm, 0, As[0], tid);
    load_tile_g2l(Bb, K, bn, 0, Bs[0], tid);
    async_stage_wait();
    __syncthreads();

    const int nk = K / TILE_K;
    for (int kt = 0; kt < nk; ++kt) {
        const int cur = kt & 1;
        const int nxt = cur ^ 1;

        if (kt + 1 < nk) {
            load_tile_g2l(Ab, K, bm, (kt + 1) * TILE_K, As[nxt], tid);
            load_tile_g2l(Bb, K, bn, (kt + 1) * TILE_K, Bs[nxt], tid);
        }
        if (kt + 2 < nk) {
            // Software prefetch of the K+2 tiles (emits global_prefetch_b8)
            __builtin_prefetch(Ab + (size_t)(bm + (tid >> 1)) * K + (size_t)(kt + 2) * TILE_K, 0, 0);
            __builtin_prefetch(Bb + (size_t)(bn + (tid >> 1)) * K + (size_t)(kt + 2) * TILE_K, 0, 0);
        }

        Frag af[4], bfr[2];
#pragma unroll
        for (int i = 0; i < 4; ++i) load_frag_lds(As[cur], wm * 64 + i * 16, lane, af[i]);
#pragma unroll
        for (int j = 0; j < 2; ++j) load_frag_lds(Bs[cur], wn * 32 + j * 16, lane, bfr[j]);

#pragma unroll
        for (int i = 0; i < 4; ++i)
#pragma unroll
            for (int j = 0; j < 2; ++j)
                acc[i][j] = __builtin_amdgcn_wmma_f32_16x16x32_bf16(
                    false, af[i].v, false, bfr[j].v, (short)0, acc[i][j], false, false);

        // All waves' async stores into the 'nxt' buffers must land before the
        // next iteration reads them: own-wave ASYNCcnt drain, then barrier.
        async_stage_wait();
        __syncthreads();
    }

    // Epilogue. C frag layout: lane n = lane&15; VGPR r -> row r + 8*(lane>>4).
    const int rsel  = (lane >> 4) * 8;
    const int nlane = lane & 15;
#pragma unroll
    for (int i = 0; i < 4; ++i) {
        const int row0 = bm + wm * 64 + i * 16 + rsel;
#pragma unroll
        for (int j = 0; j < 2; ++j) {
            const int col = bn + wn * 32 + j * 16 + nlane;
            const float bv = bias ? bias[col] : 0.0f;
#pragma unroll
            for (int r = 0; r < 8; ++r) {
                float v = acc[i][j][r] + bv;
                if (ACT == 1) v = v / (1.0f + __expf(-1.702f * v));   // GELU2
                const size_t off = (size_t)bz * sC + (size_t)(row0 + r) * N + col;
                if (RESID) v += resid[off];
                if (OUTBF) ((bf16_t*)Out)[off] = (bf16_t)v;
                else       ((float*)Out)[off]  = v;
            }
        }
    }
}

// ---------------------------------------------------------------------------
// f32 -> bf16 conversion (grid-stride)
// ---------------------------------------------------------------------------
__global__ __launch_bounds__(256)
void cvt_f32_bf16_kernel(const float* __restrict__ in, bf16_t* __restrict__ out, size_t n)
{
    for (size_t i = blockIdx.x * (size_t)blockDim.x + threadIdx.x; i < n;
         i += (size_t)gridDim.x * blockDim.x)
        out[i] = (bf16_t)in[i];
}

// ---------------------------------------------------------------------------
// Conditional LayerNorm: out = (x-mean)*rstd*(1+label@sw^T) + label@bw^T  (bf16)
// one block per (s, b) row of D=768
// ---------------------------------------------------------------------------
__global__ __launch_bounds__(256)
void cond_ln_kernel(const float* __restrict__ h, const float* __restrict__ label,
                    const float* __restrict__ sw, const float* __restrict__ bw,
                    bf16_t* __restrict__ out)
{
    const int s = blockIdx.x, b = blockIdx.y;
    const size_t base = ((size_t)b * SS + s) * DD;
    __shared__ float rsum[256], rsq[256], lab[CLS];

    float sum = 0.f, sq = 0.f;
    for (int d = threadIdx.x; d < DD; d += 256) {
        float v = h[base + d];
        sum += v; sq += v * v;
    }
    rsum[threadIdx.x] = sum;
    rsq[threadIdx.x]  = sq;
    if (threadIdx.x < CLS) lab[threadIdx.x] = label[b * CLS + threadIdx.x];
    __syncthreads();
    for (int o = 128; o > 0; o >>= 1) {
        if (threadIdx.x < o) {
            rsum[threadIdx.x] += rsum[threadIdx.x + o];
            rsq[threadIdx.x]  += rsq[threadIdx.x + o];
        }
        __syncthreads();
    }
    const float mean = rsum[0] * (1.0f / DD);
    const float var  = rsq[0] * (1.0f / DD) - mean * mean;
    const float rstd = rsqrtf(var + 1e-6f);

    for (int d = threadIdx.x; d < DD; d += 256) {
        float sc = 0.f, bi = 0.f;
#pragma unroll
        for (int c = 0; c < CLS; ++c) {
            sc += lab[c] * sw[d * CLS + c];
            bi += lab[c] * bw[d * CLS + c];
        }
        float v = (h[base + d] - mean) * rstd * (1.0f + sc) + bi;
        out[base + d] = (bf16_t)v;
    }
}

// ---------------------------------------------------------------------------
// QKV split (per-head [q|k|v] layout) + q scale -> bf16
// ---------------------------------------------------------------------------
__global__ __launch_bounds__(256)
void split_qkv_kernel(const float* __restrict__ qkv,
                      bf16_t* __restrict__ q, bf16_t* __restrict__ k,
                      bf16_t* __restrict__ v, float qscale)
{
    const size_t i = blockIdx.x * (size_t)256 + threadIdx.x;  // over B*S*D
    if (i >= (size_t)BS * DD) return;
    const int    d  = (int)(i % DD);
    const size_t bs = i / DD;
    const int    hh = d >> 6;     // head
    const int    j  = d & 63;     // within head
    const float* src = qkv + bs * DQKV + hh * 192 + j;
    q[i] = (bf16_t)(src[0]  * qscale);
    k[i] = (bf16_t)(src[64]);
    v[i] = (bf16_t)(src[128]);
}

// ---------------------------------------------------------------------------
// Causal softmax over scores[b, q, :] ; zero-fill k>q -> bf16 probs
// ---------------------------------------------------------------------------
__global__ __launch_bounds__(256)
void softmax_causal_kernel(const float* __restrict__ scores, bf16_t* __restrict__ probs)
{
    const int q = blockIdx.x, b = blockIdx.y;
    const size_t base = ((size_t)b * SS + q) * SS;
    const int n = q + 1;
    __shared__ float red[256];

    float mx = -3.4e38f;
    for (int k = threadIdx.x; k < n; k += 256) mx = fmaxf(mx, scores[base + k]);
    red[threadIdx.x] = mx;
    __syncthreads();
    for (int o = 128; o > 0; o >>= 1) {
        if (threadIdx.x < o) red[threadIdx.x] = fmaxf(red[threadIdx.x], red[threadIdx.x + o]);
        __syncthreads();
    }
    mx = red[0];
    __syncthreads();

    float s = 0.f;
    for (int k = threadIdx.x; k < n; k += 256) s += __expf(scores[base + k] - mx);
    red[threadIdx.x] = s;
    __syncthreads();
    for (int o = 128; o > 0; o >>= 1) {
        if (threadIdx.x < o) red[threadIdx.x] += red[threadIdx.x + o];
        __syncthreads();
    }
    const float inv = 1.0f / red[0];

    for (int k = threadIdx.x; k < SS; k += 256)
        probs[base + k] = (bf16_t)((k < n) ? __expf(scores[base + k] - mx) * inv : 0.0f);
}

// ---------------------------------------------------------------------------
// bf16 transpose: in[b, R, C] -> out[b, C, R]
// ---------------------------------------------------------------------------
__global__ __launch_bounds__(256)
void transpose_bf16_kernel(const bf16_t* __restrict__ in, bf16_t* __restrict__ out,
                           int R, int C)
{
    __shared__ bf16_t t[32][33];
    const int b = blockIdx.z;
    const bf16_t* src = in  + (size_t)b * R * C;
    bf16_t*       dst = out + (size_t)b * R * C;
    const int c0 = blockIdx.x * 32, r0 = blockIdx.y * 32;
    const int x = threadIdx.x, y = threadIdx.y;   // block (32, 8)
#pragma unroll
    for (int i = 0; i < 32; i += 8)
        t[y + i][x] = src[(size_t)(r0 + y + i) * C + (c0 + x)];
    __syncthreads();
#pragma unroll
    for (int i = 0; i < 32; i += 8)
        dst[(size_t)(c0 + y + i) * R + (r0 + x)] = t[x][y + i];
}

// ---------------------------------------------------------------------------
// Embed: right-shift with SOS + broadcast positional bias
// ---------------------------------------------------------------------------
__global__ __launch_bounds__(256)
void embed_kernel(const float* __restrict__ dense, const float* __restrict__ sos,
                  const float* __restrict__ p0, const float* __restrict__ p1,
                  const float* __restrict__ p2, float* __restrict__ h)
{
    const size_t i = blockIdx.x * (size_t)256 + threadIdx.x;  // over B*S*D
    if (i >= (size_t)BS * DD) return;
    const int    d  = (int)(i % DD);
    const size_t bs = i / DD;
    const int    s  = (int)(bs % SS);
    const int    b  = (int)(bs / SS);
    float pb;
    if (d < 256)      pb = p0[(s >> 8) * 256 + d];                // shape dim 0 (4)
    else if (d < 512) pb = p1[((s >> 4) & 15) * 256 + (d - 256)]; // dim 1 (16)
    else              pb = p2[(s & 15) * 256 + (d - 512)];        // dim 2 (16)
    const float base = (s == 0) ? sos[d]
                                : dense[((size_t)b * SS + (s - 1)) * DD + d];
    h[i] = base + pb;
}

// ---------------------------------------------------------------------------
// Host-side launch helpers
// ---------------------------------------------------------------------------
static void launch_gemm(hipStream_t st, const bf16_t* A, const bf16_t* Bt,
                        const float* bias, const float* resid, void* out,
                        int M, int N, int K, int batch,
                        long long sA, long long sB, long long sC,
                        int act, bool outbf)
{
    dim3 g(N / TILE_N, M / TILE_M, batch), blk(256);
    if (resid)
        gemm_bt_kernel<0, true,  false><<<g, blk, 0, st>>>(A, Bt, bias, resid, out, M, N, K, sA, sB, sC);
    else if (act == 1)
        gemm_bt_kernel<1, false, true ><<<g, blk, 0, st>>>(A, Bt, bias, nullptr, out, M, N, K, sA, sB, sC);
    else if (outbf)
        gemm_bt_kernel<0, false, true ><<<g, blk, 0, st>>>(A, Bt, bias, nullptr, out, M, N, K, sA, sB, sC);
    else
        gemm_bt_kernel<0, false, false><<<g, blk, 0, st>>>(A, Bt, bias, nullptr, out, M, N, K, sA, sB, sC);
}

static void cvt(hipStream_t st, const float* in, bf16_t* out, size_t n)
{
    cvt_f32_bf16_kernel<<<dim3(2048), dim3(256), 0, st>>>(in, out, n);
}

// ---------------------------------------------------------------------------
// Orchestration
// ---------------------------------------------------------------------------
extern "C" void kernel_launch(void* const* d_in, const int* in_sizes, int n_in,
                              void* d_out, int out_size, void* d_ws, size_t ws_size,
                              hipStream_t stream)
{
    (void)in_sizes; (void)n_in; (void)out_size; (void)ws_size;

    const float* x     = (const float*)d_in[0];
    /* d_in[1] = mask : causal, hardcoded in softmax */
    const float* label = (const float*)d_in[2];
    const float* dwi   = (const float*)d_in[3];
    const float* dbi   = (const float*)d_in[4];
    const float* sos   = (const float*)d_in[5];
    const float* pos0  = (const float*)d_in[6];
    const float* pos1  = (const float*)d_in[7];
    const float* pos2  = (const float*)d_in[8];
    const float* ln1sw = (const float*)d_in[9];
    const float* ln1bw = (const float*)d_in[10];
    const float* qkvw  = (const float*)d_in[11];
    const float* qkvb  = (const float*)d_in[12];
    const float* ao1w  = (const float*)d_in[13];
    const float* ao1b  = (const float*)d_in[14];
    const float* ao2w  = (const float*)d_in[15];
    const float* ao2b  = (const float*)d_in[16];
    const float* ln2sw = (const float*)d_in[17];
    const float* ln2bw = (const float*)d_in[18];
    const float* w1    = (const float*)d_in[19];
    const float* b1    = (const float*)d_in[20];
    const float* w2    = (const float*)d_in[21];
    const float* b2    = (const float*)d_in[22];
    const float* lnfsw = (const float*)d_in[23];
    const float* lnfbw = (const float*)d_in[24];
    const float* outw  = (const float*)d_in[25];
    const float* outb  = (const float*)d_in[26];

    // Workspace carving (256B aligned)
    char* p = (char*)d_ws;
    auto carve = [&](size_t bytes) -> char* {
        char* r = p;
        p += (bytes + 255) & ~(size_t)255;
        return r;
    };
    float*  h     = (float*) carve((size_t)BS * DD * 4);          // residual stream
    float*  big   = (float*) carve((size_t)BS * DQKV * 4);        // qkv / scores / dense scratch
    bf16_t* bfa   = (bf16_t*)carve((size_t)BS * DD * 2);          // ln out / attn out
    bf16_t* bfb   = (bf16_t*)carve((size_t)BS * DD * 2);          // o1 out
    bf16_t* qbf   = (bf16_t*)carve((size_t)BS * DD * 2);
    bf16_t* kbf   = (bf16_t*)carve((size_t)BS * DD * 2);
    bf16_t* vbf   = (bf16_t*)carve((size_t)BS * DD * 2);
    bf16_t* vt    = (bf16_t*)carve((size_t)BS * DD * 2);
    bf16_t* probs = (bf16_t*)carve((size_t)BB * SS * SS * 2);
    bf16_t* mbf   = (bf16_t*)carve((size_t)BS * FF * 2);
    bf16_t* wbuf  = (bf16_t*)carve((size_t)FF * DD * 2);          // largest weight slice
    bf16_t* xbf   = (bf16_t*)carve((size_t)BS * 64 * 2);

    const int elemBlocks = (int)(((size_t)BS * DD + 255) / 256);
    const float qscale = 1.0f / sqrtf((float)DD);   // module quirk: full D, not head_dim

    // ---- input embedding: dense_in GEMM -> shift+SOS+posbias ----
    cvt(stream, x, xbf, (size_t)BS * 64);
    cvt(stream, dwi, wbuf, (size_t)DD * 64);
    launch_gemm(stream, xbf, wbuf, dbi, nullptr, big, BS, DD, 64, 1, 0, 0, 0, 0, false);
    embed_kernel<<<dim3(elemBlocks), dim3(256), 0, stream>>>(big, sos, pos0, pos1, pos2, h);

    // ---- transformer layers ----
    for (int l = 0; l < NL; ++l) {
        // cond LN 1
        cond_ln_kernel<<<dim3(SS, BB), dim3(256), 0, stream>>>(
            h, label, ln1sw + (size_t)l * DD * CLS, ln1bw + (size_t)l * DD * CLS, bfa);

        // QKV projection
        cvt(stream, qkvw + (size_t)l * DQKV * DD, wbuf, (size_t)DQKV * DD);
        launch_gemm(stream, bfa, wbuf, qkvb + (size_t)l * DQKV, nullptr, big,
                    BS, DQKV, DD, 1, 0, 0, 0, 0, false);
        split_qkv_kernel<<<dim3(elemBlocks), dim3(256), 0, stream>>>(big, qbf, kbf, vbf, qscale);

        // scores = (q * D^-0.5) @ k^T  (batched over B)
        launch_gemm(stream, qbf, kbf, nullptr, nullptr, big,
                    SS, SS, DD, BB,
                    (long long)SS * DD, (long long)SS * DD, (long long)SS * SS, 0, false);
        softmax_causal_kernel<<<dim3(SS, BB), dim3(256), 0, stream>>>(big, probs);

        // attn out = probs @ v   (via v^T so the A·B^T GEMM is reused)
        transpose_bf16_kernel<<<dim3(DD / 32, SS / 32, BB), dim3(32, 8), 0, stream>>>(vbf, vt, SS, DD);
        launch_gemm(stream, probs, vt, nullptr, nullptr, bfa,
                    SS, DD, SS, BB,
                    (long long)SS * SS, (long long)DD * SS, (long long)SS * DD, 0, true);

        // two output projections; residual folded into the second
        cvt(stream, ao1w + (size_t)l * DD * DD, wbuf, (size_t)DD * DD);
        launch_gemm(stream, bfa, wbuf, ao1b + (size_t)l * DD, nullptr, bfb,
                    BS, DD, DD, 1, 0, 0, 0, 0, true);
        cvt(stream, ao2w + (size_t)l * DD * DD, wbuf, (size_t)DD * DD);
        launch_gemm(stream, bfb, wbuf, ao2b + (size_t)l * DD, h, h,
                    BS, DD, DD, 1, 0, 0, 0, 0, false);

        // cond LN 2 + MLP (GELU2 fused into mlp1 epilogue, residual into mlp2)
        cond_ln_kernel<<<dim3(SS, BB), dim3(256), 0, stream>>>(
            h, label, ln2sw + (size_t)l * DD * CLS, ln2bw + (size_t)l * DD * CLS, bfa);
        cvt(stream, w1 + (size_t)l * FF * DD, wbuf, (size_t)FF * DD);
        launch_gemm(stream, bfa, wbuf, b1 + (size_t)l * FF, nullptr, mbf,
                    BS, FF, DD, 1, 0, 0, 0, 1, true);
        cvt(stream, w2 + (size_t)l * DD * FF, wbuf, (size_t)DD * FF);
        launch_gemm(stream, mbf, wbuf, b2 + (size_t)l * DD, h, h,
                    BS, DD, FF, 1, 0, 0, 0, 0, false);
    }

    // ---- final LN + output projection ----
    cond_ln_kernel<<<dim3(SS, BB), dim3(256), 0, stream>>>(h, label, lnfsw, lnfbw, bfa);
    cvt(stream, outw, wbuf, (size_t)OUTD * DD);
    launch_gemm(stream, bfa, wbuf, outb, nullptr, (float*)d_out,
                BS, OUTD, DD, 1, 0, 0, 0, 0, false);
}
